// PointTransformer_43782896615731
// MI455X (gfx1250) — matrix operand order repaired and checked by reading, not compiled, not run on hardware
//
#include <hip/hip_runtime.h>
#include <cstddef>
#include <cstdint>

#define BB   4      // batch
#define KNN  16
#define BN_EPS 1e-5f

typedef __attribute__((ext_vector_type(16))) __bf16 bf16x16;
typedef __attribute__((ext_vector_type(8)))  float  f32x8;
typedef __attribute__((ext_vector_type(4)))  float  f32x4;

// ============================================================================
// Generic GEMM: Y[M,N] = act( X[M,K] @ W[N,K]^T + bias (+ residual) )
// Requirements (guaranteed by all call sites): M % 16 == 0, N % 32 == 0.
// One wave32 computes a 16x32 tile (two 16x16 WMMA tiles sharing the A frag)
// with v_wmma_f32_16x16x32_bf16; f32 accumulation.
// Fast path (K%4==0, rows 16B-aligned): 12 independent float4 loads per
// K-slice -> one load clause, overlapped waits. Odd-K path: scalar loads.
// Final partial K-slice (K%32!=0) takes a guarded path once.
// ============================================================================
__global__ __launch_bounds__(256)
void k_gemm_wmma(const float* __restrict__ X, const float* __restrict__ W,
                 const float* __restrict__ bias, const float* __restrict__ res,
                 float* __restrict__ Y, int M, int K, int N, int relu)
{
  const int lane = threadIdx.x & 31;
  const int wave = threadIdx.x >> 5;
  const int TN2 = N >> 5;                // 16x32 tiles along N
  const int TM  = M >> 4;
  const int t   = blockIdx.x * 8 + wave;
  if (t >= TM * TN2) return;
  const int mt = t / TN2, nt = t % TN2;
  const int m0 = mt << 4, n0 = nt << 5;
  const int lh = lane >> 4;              // lane half (0/1)
  const int ll = lane & 15;

  // A lane layout (16x32 bf16): m = ll; j<4 -> k = 8*lh+2j ; j>=4 -> 8*lh+16+2(j-4)
  // B lane layout (32x16 bf16): n = ll; same k order as A
  const float* xrow  = X + (size_t)(m0 + ll) * K;
  const float* w0row = W + (size_t)(n0 + ll) * K;
  const float* w1row = W + (size_t)(n0 + 16 + ll) * K;

  f32x8 acc0 = {0.f, 0.f, 0.f, 0.f, 0.f, 0.f, 0.f, 0.f};
  f32x8 acc1 = {0.f, 0.f, 0.f, 0.f, 0.f, 0.f, 0.f, 0.f};

  const int Kmain = K & ~31;
  int k0 = 0;

  if ((K & 3) == 0) {
    // ---- fast path: rows 16B-aligned, wide loads into independent regs ----
    for (; k0 < Kmain; k0 += 32) {
      __builtin_prefetch(xrow + k0 + 32);       // speculative; OOB is dropped
      __builtin_prefetch(w0row + k0 + 32);
      __builtin_prefetch(w1row + k0 + 32);

      const f32x4* ax  = (const f32x4*)(xrow  + k0 + 8 * lh);   // 8-float aligned
      const f32x4* bx0 = (const f32x4*)(w0row + k0 + 8 * lh);   // 8-float aligned
      const f32x4* bx1 = (const f32x4*)(w1row + k0 + 8 * lh);
      // 12 independent b128 loads -> one clause, overlapped latency
      const f32x4 a0 = ax[0],  a1 = ax[1];      // K = 8*lh + 0..7
      const f32x4 a2 = ax[4],  a3 = ax[5];      // K = 8*lh + 16..23
      const f32x4 p0 = bx0[0], p1 = bx0[1], p2 = bx0[4], p3 = bx0[5];
      const f32x4 q0 = bx1[0], q1 = bx1[1], q2 = bx1[4], q3 = bx1[5];

      bf16x16 a, b0, b1;
#pragma unroll
      for (int j = 0; j < 4; ++j) {
        a[j]      = static_cast<__bf16>(a0[j]);
        a[4 + j]  = static_cast<__bf16>(a1[j]);
        a[8 + j]  = static_cast<__bf16>(a2[j]);
        a[12 + j] = static_cast<__bf16>(a3[j]);
        b0[j]      = static_cast<__bf16>(p0[j]);
        b0[4 + j]  = static_cast<__bf16>(p1[j]);
        b0[8 + j]  = static_cast<__bf16>(p2[j]);
        b0[12 + j] = static_cast<__bf16>(p3[j]);
        b1[j]      = static_cast<__bf16>(q0[j]);
        b1[4 + j]  = static_cast<__bf16>(q1[j]);
        b1[8 + j]  = static_cast<__bf16>(q2[j]);
        b1[12 + j] = static_cast<__bf16>(q3[j]);
      }
      acc0 = __builtin_amdgcn_wmma_f32_16x16x32_bf16(false, a, false, b0,
                                                     (short)0, acc0, false, false);
      acc1 = __builtin_amdgcn_wmma_f32_16x16x32_bf16(false, a, false, b1,
                                                     (short)0, acc1, false, false);
    }
  } else {
    // ---- generic path: scalar loads (rows only 4B-aligned) ----
    for (; k0 < Kmain; k0 += 32) {
      __builtin_prefetch(xrow + k0 + 32);
      __builtin_prefetch(w0row + k0 + 32);
      __builtin_prefetch(w1row + k0 + 32);

      bf16x16 a, b0, b1;
      const float* ax = xrow + k0 + 8 * lh;
#pragma unroll
      for (int j = 0; j < 4; ++j) {             // K = 8*lh + 0..7
        a[2 * j]     = static_cast<__bf16>(ax[2 * j]);
        a[2 * j + 1] = static_cast<__bf16>(ax[2 * j + 1]);
      }
#pragma unroll
      for (int j = 0; j < 4; ++j) {             // K = 8*lh + 16..23
        a[8 + 2 * j] = static_cast<__bf16>(ax[16 + 2 * j]);
        a[9 + 2 * j] = static_cast<__bf16>(ax[17 + 2 * j]);
      }
      const float* bw0 = w0row + k0 + 8 * lh;
      const float* bw1 = w1row + k0 + 8 * lh;
#pragma unroll
      for (int j = 0; j < 8; ++j) {             // K = 8*lh + 0..7, 8*lh + 16..23
        const int kk = 2 * j + ((j < 4) ? 0 : 8);
        b0[2 * j]     = static_cast<__bf16>(bw0[kk]);
        b0[2 * j + 1] = static_cast<__bf16>(bw0[kk + 1]);
        b1[2 * j]     = static_cast<__bf16>(bw1[kk]);
        b1[2 * j + 1] = static_cast<__bf16>(bw1[kk + 1]);
      }
      acc0 = __builtin_amdgcn_wmma_f32_16x16x32_bf16(false, a, false, b0,
                                                     (short)0, acc0, false, false);
      acc1 = __builtin_amdgcn_wmma_f32_16x16x32_bf16(false, a, false, b1,
                                                     (short)0, acc1, false, false);
    }
  }

  if (k0 < K) {                                  // guarded tail (once at most)
    bf16x16 a, b0, b1;
#pragma unroll
    for (int j = 0; j < 8; ++j) {
      const int k = k0 + 8 * lh + ((j < 4) ? (2 * j) : (16 + 2 * (j - 4)));
      a[2 * j]     = static_cast<__bf16>((k     < K) ? xrow[k]     : 0.f);
      a[2 * j + 1] = static_cast<__bf16>((k + 1 < K) ? xrow[k + 1] : 0.f);
    }
#pragma unroll
    for (int j = 0; j < 8; ++j) {
      const int k = k0 + 8 * lh + ((j < 4) ? (2 * j) : (16 + 2 * (j - 4)));
      b0[2 * j]     = static_cast<__bf16>((k     < K) ? w0row[k]     : 0.f);
      b0[2 * j + 1] = static_cast<__bf16>((k + 1 < K) ? w0row[k + 1] : 0.f);
      b1[2 * j]     = static_cast<__bf16>((k     < K) ? w1row[k]     : 0.f);
      b1[2 * j + 1] = static_cast<__bf16>((k + 1 < K) ? w1row[k + 1] : 0.f);
    }
    acc0 = __builtin_amdgcn_wmma_f32_16x16x32_bf16(false, a, false, b0,
                                                   (short)0, acc0, false, false);
    acc1 = __builtin_amdgcn_wmma_f32_16x16x32_bf16(false, a, false, b1,
                                                   (short)0, acc1, false, false);
  }

  // D layout: VGPR r -> m = m0 + r + 8*lh, n = ll (per 16-wide tile)
  const float bv0 = bias ? bias[n0 + ll]      : 0.f;
  const float bv1 = bias ? bias[n0 + 16 + ll] : 0.f;
#pragma unroll
  for (int r = 0; r < 8; ++r) {
    const int m = m0 + r + (lh << 3);
    const size_t o0 = (size_t)m * N + n0 + ll;
    const size_t o1 = o0 + 16;
    float v0 = acc0[r] + bv0;
    float v1 = acc1[r] + bv1;
    if (res)  { v0 += res[o0]; v1 += res[o1]; }
    if (relu) { v0 = fmaxf(v0, 0.f); v1 = fmaxf(v1, 0.f); }
    Y[o0] = v0;
    Y[o1] = v1;
  }
}

// ============================================================================
// Fused KNN (top-16 smallest sq-distances), never materializing the NxN matrix.
// Source points streamed through LDS tiles of 256.
// ============================================================================
__global__ __launch_bounds__(128)
void k_knn(const float* __restrict__ query, const float* __restrict__ src,
           int* __restrict__ out, int Mq, int Ns)
{
  __shared__ float sx[256], sy[256], sz[256];
  const int b = blockIdx.y;
  const int m = blockIdx.x * 128 + threadIdx.x;
  const bool active = (m < Mq);
  float qx = 0.f, qy = 0.f, qz = 0.f;
  if (active) {
    const float* qp = query + ((size_t)b * Mq + m) * 3;
    qx = qp[0]; qy = qp[1]; qz = qp[2];
  }
  float bd[KNN]; int bi[KNN];
#pragma unroll
  for (int i = 0; i < KNN; ++i) { bd[i] = 3.0e38f; bi[i] = 0; }

  for (int base = 0; base < Ns; base += 256) {
    const int cnt = (Ns - base < 256) ? (Ns - base) : 256;
    for (int j = threadIdx.x; j < cnt; j += 128) {
      const float* sp = src + ((size_t)b * Ns + base + j) * 3;
      sx[j] = sp[0]; sy[j] = sp[1]; sz[j] = sp[2];
    }
    __syncthreads();
    if (active) {
      for (int j = 0; j < cnt; ++j) {
        const float dx = qx - sx[j], dy = qy - sy[j], dz = qz - sz[j];
        const float d = dx * dx + dy * dy + dz * dz;
        if (d < bd[KNN - 1]) {
          bd[KNN - 1] = d; bi[KNN - 1] = base + j;
#pragma unroll
          for (int p = KNN - 1; p > 0; --p) {
            if (bd[p] < bd[p - 1]) {
              const float tf_ = bd[p]; bd[p] = bd[p - 1]; bd[p - 1] = tf_;
              const int   ti_ = bi[p]; bi[p] = bi[p - 1]; bi[p - 1] = ti_;
            }
          }
        }
      }
    }
    __syncthreads();
  }
  if (active) {
    int* o = out + ((size_t)b * Mq + m) * KNN;
#pragma unroll
    for (int i = 0; i < KNN; ++i) o[i] = bi[i];
  }
}

// ============================================================================
// Farthest point sampling: one block per batch; dist[] lives in LDS (16 KB).
// Matches reference scan semantics: out[0]=0; out[t]=argmax after t updates.
// ============================================================================
__global__ __launch_bounds__(256)
void k_fps(const float* __restrict__ xyz, int* __restrict__ out, int Ns, int Snp)
{
  const int b = blockIdx.x;
  const float* P = xyz + (size_t)b * Ns * 3;
  __shared__ float dist[4096];
  __shared__ float rmax[256];
  __shared__ int   rarg[256];
  __shared__ int   curFar;
  for (int i = threadIdx.x; i < Ns; i += 256) dist[i] = 1e10f;
  if (threadIdx.x == 0) curFar = 0;
  __syncthreads();

  for (int it = 0; it < Snp; ++it) {
    const int far = curFar;
    if (threadIdx.x == 0) out[b * Snp + it] = far;
    const float cx = P[far * 3], cy = P[far * 3 + 1], cz = P[far * 3 + 2];
    float lm = -1.f; int la = 0;
    for (int i = threadIdx.x; i < Ns; i += 256) {
      const float dx = P[i * 3] - cx, dy = P[i * 3 + 1] - cy, dz = P[i * 3 + 2] - cz;
      const float dd = dx * dx + dy * dy + dz * dz;
      const float dn = fminf(dist[i], dd);
      dist[i] = dn;
      if (dn > lm) { lm = dn; la = i; }
    }
    rmax[threadIdx.x] = lm; rarg[threadIdx.x] = la;
    __syncthreads();
    for (int s = 128; s > 0; s >>= 1) {
      if (threadIdx.x < s) {
        const float om = rmax[threadIdx.x + s]; const int oa = rarg[threadIdx.x + s];
        if (om > rmax[threadIdx.x] ||
            (om == rmax[threadIdx.x] && oa < rarg[threadIdx.x])) {
          rmax[threadIdx.x] = om; rarg[threadIdx.x] = oa;
        }
      }
      __syncthreads();
    }
    if (threadIdx.x == 0) curFar = rarg[0];
    __syncthreads();
  }
}

// ============================================================================
// Small fused element-wise kernels
// ============================================================================
__global__ void k_gather(const float* __restrict__ in, const int* __restrict__ idx,
                         float* __restrict__ out, int Ns, int S, int C, size_t total)
{
  const size_t t = (size_t)blockIdx.x * blockDim.x + threadIdx.x;
  if (t >= total) return;
  const int c  = (int)(t % C);
  const size_t bs = t / C;
  const int b = (int)(bs / S);
  out[t] = in[((size_t)b * Ns + idx[bs]) * C + c];
}

// pe_in[b,m,k,:] = pos[b,m] - pos[b, idx[b,m,k]]
__global__ void k_pe_in(const float* __restrict__ pos, const int* __restrict__ idx,
                        float* __restrict__ out, int Mq, size_t total)  // total = B*Mq*KNN
{
  const size_t t = (size_t)blockIdx.x * blockDim.x + threadIdx.x;
  if (t >= total) return;
  const size_t bm = t / KNN;
  const int b = (int)(bm / Mq);
  const int n = idx[t];
  const float* pm = pos + bm * 3;
  const float* pn = pos + ((size_t)b * Mq + n) * 3;
  float* o = out + t * 3;
  o[0] = pm[0] - pn[0]; o[1] = pm[1] - pn[1]; o[2] = pm[2] - pn[2];
}

// attn_in[b,m,k,d] = q[b,m,d] - kfeat[b, idx[b,m,k], d] + pe[b,m,k,d]
__global__ void k_attn_in(const float* __restrict__ q, const float* __restrict__ kf,
                          const float* __restrict__ pe, const int* __restrict__ idx,
                          float* __restrict__ out, int Mq, int D, size_t total)
{
  const size_t t = (size_t)blockIdx.x * blockDim.x + threadIdx.x;
  if (t >= total) return;
  const int d = (int)(t % D);
  const size_t bmk = t / D;
  const int k = (int)(bmk % KNN);
  const size_t bm = bmk / KNN;
  const int b = (int)(bm / Mq);
  const int n = idx[bm * KNN + k];
  out[t] = q[bm * D + d] - kf[((size_t)b * Mq + n) * D + d] + pe[t];
}

// per (b,m,d): softmax over 16 neighbors of attn/sqrt(D), aggregate (v_gather + pe)
__global__ void k_softmax_agg(const float* __restrict__ attn, const float* __restrict__ v,
                              const float* __restrict__ pe, const int* __restrict__ idx,
                              float* __restrict__ out, int Mq, int D, size_t total)
{
  const size_t t = (size_t)blockIdx.x * blockDim.x + threadIdx.x;
  if (t >= total) return;
  const int d = (int)(t % D);
  const size_t bm = t / D;
  const int b = (int)(bm / Mq);
  const float inv = rsqrtf((float)D);
  const int* ip = idx + bm * KNN;
  float s[KNN];
  float mx = -3.0e38f;
#pragma unroll
  for (int k = 0; k < KNN; ++k) {
    const float a = attn[(bm * KNN + k) * D + d] * inv;
    s[k] = a; mx = fmaxf(mx, a);
  }
  float den = 0.f, num = 0.f;
#pragma unroll
  for (int k = 0; k < KNN; ++k) {
    const float e = __expf(s[k] - mx);
    const int n = ip[k];
    const float val = v[((size_t)b * Mq + n) * D + d] + pe[(bm * KNN + k) * D + d];
    den += e; num += e * val;
  }
  out[bm * D + d] = num / den;
}

// g[b,s,k,c] : c<3 -> rel xyz, else gathered features
__global__ void k_build_g(const float* __restrict__ xyz, const float* __restrict__ feat,
                          const float* __restrict__ nxyz, const int* __restrict__ idx,
                          float* __restrict__ out, int S, int Ns, int C, size_t total)
{
  const size_t t = (size_t)blockIdx.x * blockDim.x + threadIdx.x;
  if (t >= total) return;
  const int ctot = C + 3;
  const int c = (int)(t % ctot);
  const size_t bsk = t / ctot;
  const size_t bs  = bsk / KNN;
  const int b = (int)(bs / S);
  const int n = idx[bsk];
  float v;
  if (c < 3) v = xyz[((size_t)b * Ns + n) * 3 + c] - nxyz[bs * 3 + c];
  else       v = feat[((size_t)b * Ns + n) * C + (c - 3)];
  out[t] = v;
}

// BN stats: LDS partials (ds_add_f32) + global atomics. stats[0..C)=sum, [C..2C)=sumsq
__global__ __launch_bounds__(256)
void k_bn_stats(const float* __restrict__ X, float* __restrict__ stats, int C, size_t total)
{
  __shared__ float ssum[512], ssq[512];
  for (int c = threadIdx.x; c < C; c += 256) { ssum[c] = 0.f; ssq[c] = 0.f; }
  __syncthreads();
  const size_t stride = (size_t)gridDim.x * 256;
  for (size_t e = (size_t)blockIdx.x * 256 + threadIdx.x; e < total; e += stride) {
    const float v = X[e];
    const int c = (int)(e % (size_t)C);
    atomicAdd(&ssum[c], v);
    atomicAdd(&ssq[c], v * v);
  }
  __syncthreads();
  for (int c = threadIdx.x; c < C; c += 256) {
    atomicAdd(&stats[c], ssum[c]);
    atomicAdd(&stats[C + c], ssq[c]);
  }
}

__global__ void k_bn_apply(float* __restrict__ y, const float* __restrict__ stats,
                           const float* __restrict__ gamma, const float* __restrict__ beta,
                           int C, float invcnt, size_t total)
{
  const size_t t = (size_t)blockIdx.x * blockDim.x + threadIdx.x;
  if (t >= total) return;
  const int c = (int)(t % C);
  const float mu  = stats[c] * invcnt;
  const float var = stats[C + c] * invcnt - mu * mu;
  const float v = (y[t] - mu) * rsqrtf(var + BN_EPS) * gamma[c] + beta[c];
  y[t] = fmaxf(v, 0.f);
}

__global__ void k_maxpool(const float* __restrict__ in, float* __restrict__ out,
                          int C, size_t total)   // in [B,S,KNN,C] -> out [B,S,C]
{
  const size_t t = (size_t)blockIdx.x * blockDim.x + threadIdx.x;
  if (t >= total) return;
  const int c = (int)(t % C);
  const size_t bs = t / C;
  float m = -3.0e38f;
#pragma unroll
  for (int k = 0; k < KNN; ++k) m = fmaxf(m, in[(bs * KNN + k) * C + c]);
  out[t] = m;
}

// ============================================================================
// Host orchestration
// ============================================================================
struct PtbP {
  const float *d1W,*d1b,*d2W,*d2b,*f1W,*f1b,*f2W,*f2b,*g1W,*g1b,*g2W,*g2b,*wkW,*wqW,*wvW;
};
struct Ctx {
  hipStream_t s;
  float *bufA,*bufB,*bufC,*h1,*q,*kf,*vf,*pein,*res;
  int   *idx;
};

static inline dim3 grid1(size_t total) { return dim3((unsigned)((total + 255) / 256)); }

static void launch_gemm(const Ctx& C, const float* X, const float* W, const float* b,
                        const float* res, float* Y, int M, int K, int N, int relu)
{
  const int tiles  = (M / 16) * (N / 32);   // M%16==0, N%32==0 at every call site
  const int blocks = (tiles + 7) / 8;
  k_gemm_wmma<<<blocks, 256, 0, C.s>>>(X, W, b, res, Y, M, K, N, relu);
}

static void run_ptb(const Ctx& C, const PtbP& P, const float* pos,
                    const float* h_in, float* h_out, int M, int D)
{
  const int BM = BB * M;
  dim3 kg((M + 127) / 128, BB);
  k_knn<<<kg, 128, 0, C.s>>>(pos, pos, C.idx, M, M);
  launch_gemm(C, h_in, P.f1W, P.f1b, nullptr, C.h1, BM, D, D, 0);          // fc1
  launch_gemm(C, C.h1, P.wqW, nullptr, nullptr, C.q,  BM, D, D, 0);        // wq
  launch_gemm(C, C.h1, P.wkW, nullptr, nullptr, C.kf, BM, D, D, 0);        // wk
  launch_gemm(C, C.h1, P.wvW, nullptr, nullptr, C.vf, BM, D, D, 0);        // wv
  const size_t tK = (size_t)BM * KNN;
  k_pe_in<<<grid1(tK), 256, 0, C.s>>>(pos, C.idx, C.pein, M, tK);
  launch_gemm(C, C.pein, P.d1W, P.d1b, nullptr, C.bufC, BM * KNN, 3, D, 1);   // delta1+relu
  launch_gemm(C, C.bufC, P.d2W, P.d2b, nullptr, C.bufB, BM * KNN, D, D, 0);   // pe = delta2
  const size_t tE = (size_t)BM * KNN * D;
  k_attn_in<<<grid1(tE), 256, 0, C.s>>>(C.q, C.kf, C.bufB, C.idx, C.bufA, M, D, tE);
  launch_gemm(C, C.bufA, P.g1W, P.g1b, nullptr, C.bufC, BM * KNN, D, D, 1);   // gamma1+relu
  launch_gemm(C, C.bufC, P.g2W, P.g2b, nullptr, C.bufA, BM * KNN, D, D, 0);   // gamma2
  const size_t tR = (size_t)BM * D;
  k_softmax_agg<<<grid1(tR), 256, 0, C.s>>>(C.bufA, C.vf, C.bufB, C.idx, C.res, M, D, tR);
  launch_gemm(C, C.res, P.f2W, P.f2b, h_in, h_out, BM, D, D, 0);           // fc2 + residual
}

static void run_sa(const Ctx& C,
                   const float* W0, const float* b0, const float* beta0, const float* gamma0,
                   const float* W1, const float* b1, const float* beta1, const float* gamma1,
                   const float* pos, const float* h_in,
                   int Ns, int S, int Dprev, int Cout,
                   float* posNext, float* h_out, int* fpsIdx, int* idx2, float* stats)
{
  k_fps<<<BB, 256, 0, C.s>>>(pos, fpsIdx, Ns, S);
  const size_t tP = (size_t)BB * S * 3;
  k_gather<<<grid1(tP), 256, 0, C.s>>>(pos, fpsIdx, posNext, Ns, S, 3, tP);
  dim3 kg((S + 127) / 128, BB);
  k_knn<<<kg, 128, 0, C.s>>>(posNext, pos, idx2, S, Ns);
  const int ctot = Dprev + 3;
  const size_t tG = (size_t)BB * S * KNN * ctot;
  k_build_g<<<grid1(tG), 256, 0, C.s>>>(pos, h_in, posNext, idx2, C.bufA, S, Ns, Dprev, tG);
  const int rows = BB * S * KNN;
  const size_t tY = (size_t)rows * Cout;
  const float invcnt = 1.0f / (float)rows;
  // layer 0
  launch_gemm(C, C.bufA, W0, b0, nullptr, C.bufB, rows, ctot, Cout, 0);
  hipMemsetAsync(stats, 0, 1024 * sizeof(float), C.s);
  k_bn_stats<<<512, 256, 0, C.s>>>(C.bufB, stats, Cout, tY);
  k_bn_apply<<<grid1(tY), 256, 0, C.s>>>(C.bufB, stats, gamma0, beta0, Cout, invcnt, tY);
  // layer 1
  launch_gemm(C, C.bufB, W1, b1, nullptr, C.bufC, rows, Cout, Cout, 0);
  hipMemsetAsync(stats, 0, 1024 * sizeof(float), C.s);
  k_bn_stats<<<512, 256, 0, C.s>>>(C.bufC, stats, Cout, tY);
  k_bn_apply<<<grid1(tY), 256, 0, C.s>>>(C.bufC, stats, gamma1, beta1, Cout, invcnt, tY);
  // maxpool over K
  const size_t tO = (size_t)BB * S * Cout;
  k_maxpool<<<grid1(tO), 256, 0, C.s>>>(C.bufC, h_out, Cout, tO);
}

extern "C" void kernel_launch(void* const* d_in, const int* in_sizes, int n_in,
                              void* d_out, int out_size, void* d_ws, size_t ws_size,
                              hipStream_t stream)
{
  (void)in_sizes; (void)n_in; (void)out_size; (void)ws_size;
  // ---- parse inputs (jax pytree order: dicts sorted by key, lists in order) ----
  // top: "params" < "x"; params: fc, ptb, stages; linear dict: W < b; sa layer: W,b,beta,gamma
  int ci = 0;
  auto nxt = [&]() { return (const float*)d_in[ci++]; };
  const float* fcl1W = nxt(); const float* fcl1b = nxt();
  const float* fcl2W = nxt(); const float* fcl2b = nxt();
  auto parse_ptb = [&]() {
    PtbP p;  // sorted: delta1, delta2, fc1, fc2, gamma1, gamma2, wk, wq, wv
    p.d1W = nxt(); p.d1b = nxt(); p.d2W = nxt(); p.d2b = nxt();
    p.f1W = nxt(); p.f1b = nxt(); p.f2W = nxt(); p.f2b = nxt();
    p.g1W = nxt(); p.g1b = nxt(); p.g2W = nxt(); p.g2b = nxt();
    p.wkW = nxt(); p.wqW = nxt(); p.wvW = nxt();
    return p;
  };
  PtbP ptb0 = parse_ptb();
  struct SaL { const float *W, *b, *beta, *gamma; };
  SaL sa[4][2]; PtbP tf[4];
  for (int s = 0; s < 4; ++s) {
    for (int l = 0; l < 2; ++l) {
      sa[s][l].W = nxt(); sa[s][l].b = nxt(); sa[s][l].beta = nxt(); sa[s][l].gamma = nxt();
    }
    tf[s] = parse_ptb();
  }
  const float* x = nxt();   // [B,4096,3]

  // ---- workspace layout (floats, 256B aligned) ----
  size_t off = 0;
  auto alloc = [&](size_t n) {
    float* p = (float*)d_ws + off;
    off += (n + 63) & ~(size_t)63;
    return p;
  };
  Ctx C; C.s = stream;
  const size_t BIG = (size_t)BB * 4096 * KNN * 32;   // 8,388,608 floats = 32 MB
  C.bufA = alloc(BIG); C.bufB = alloc(BIG); C.bufC = alloc(BIG);
  C.h1   = alloc((size_t)BB * 4096 * 32);
  C.q    = alloc((size_t)BB * 4096 * 32);
  C.kf   = alloc((size_t)BB * 4096 * 32);
  C.vf   = alloc((size_t)BB * 4096 * 32);
  C.pein = alloc((size_t)BB * 4096 * KNN * 3);
  C.res  = alloc((size_t)BB * 4096 * 32);
  C.idx  = (int*)alloc((size_t)BB * 4096 * KNN);
  float* hA    = alloc((size_t)BB * 4096 * 32);
  float* hB    = alloc((size_t)BB * 4096 * 32);
  float* posb0 = alloc((size_t)BB * 1024 * 3);
  float* posb1 = alloc((size_t)BB * 1024 * 3);
  int*   fpsIdx = (int*)alloc((size_t)BB * 1024);
  int*   idx2   = (int*)alloc((size_t)BB * 1024 * KNN);
  float* stats  = alloc(1024);

  // ---- initial MLP: h = l2(relu(l1(x))) ----
  launch_gemm(C, x, fcl1W, fcl1b, nullptr, C.bufC, BB * 4096, 3, 32, 1);
  launch_gemm(C, C.bufC, fcl2W, fcl2b, nullptr, hA, BB * 4096, 32, 32, 0);
  run_ptb(C, ptb0, x, hA, hB, 4096, 32);

  static const int NPT[4] = {1024, 256, 64, 16};
  static const int BHD[4] = {64, 128, 256, 512};
  const float* posCur = x;
  float* hCur = hB;
  int Ns = 4096, Dprev = 32;
  for (int s = 0; s < 4; ++s) {
    float* posNext = (s & 1) ? posb1 : posb0;
    float* hNext = (hCur == hA) ? hB : hA;
    run_sa(C, sa[s][0].W, sa[s][0].b, sa[s][0].beta, sa[s][0].gamma,
              sa[s][1].W, sa[s][1].b, sa[s][1].beta, sa[s][1].gamma,
           posCur, hCur, Ns, NPT[s], Dprev, BHD[s],
           posNext, hNext, fpsIdx, idx2, stats);
    float* hOut = (s == 3) ? (float*)d_out : ((hNext == hA) ? hB : hA);
    run_ptb(C, tf[s], posNext, hNext, hOut, NPT[s], BHD[s]);
    posCur = posNext; hCur = hOut; Ns = NPT[s]; Dprev = BHD[s];
  }
}
